// Translation_8735963480767
// MI455X (gfx1250) — matrix operand order repaired and checked by reading, not compiled, not run on hardware
//
#include <hip/hip_runtime.h>
#include <math.h>

// ---------------- model dims ----------------
#define BB    128          // batch
#define SS    256          // sequence length (enc steps == dec steps)
#define HH    512          // hidden
#define SRCN  300
#define SRCP  320          // SRC padded to mult of 32 (WMMA K)
#define TGTN  300
#define TGTP  320          // TGT padded to mult of 64 (full N tiles -> branchless K loop)
#define G4H   2048         // 4*H
#define H2    1024         // 2*H

#define BROW  40           // LDS B-tile row stride in elements (32 + 8 pad, bank-spread)

typedef __bf16 bf16_t;
typedef __attribute__((ext_vector_type(16))) __bf16 v16bf;
typedef __attribute__((ext_vector_type(8)))  float  v8f;
typedef __attribute__((ext_vector_type(4)))  unsigned int v4u;

union ABFrag { v16bf v; v4u q[2]; };

__device__ __forceinline__ bf16_t f2bf(float x) {
  unsigned u = __builtin_bit_cast(unsigned, x);
  unsigned r = u + 0x7fffu + ((u >> 16) & 1u);          // round-to-nearest-even
  unsigned short h = (unsigned short)(r >> 16);
  return __builtin_bit_cast(bf16_t, h);
}
__device__ __forceinline__ float bf2f(bf16_t b) {
  unsigned short h = __builtin_bit_cast(unsigned short, b);
  unsigned u = ((unsigned)h) << 16;
  return __builtin_bit_cast(float, u);
}
__device__ __forceinline__ float sigm(float x) { return 1.0f / (1.0f + expf(-x)); }

// A fragment: 16x32 bf16, row-major source, stride lda.
// lane<16: M=lane, K = {k0..k0+7, k0+16..k0+23}; lane>=16: +8 on K base.
__device__ __forceinline__ v16bf load_frag_A(const bf16_t* A, int lda, int mrow, int k0, int hf) {
  const bf16_t* p = A + (size_t)mrow * lda + k0 + hf * 8;
  ABFrag f;
  f.q[0] = *(const v4u*)p;
  f.q[1] = *(const v4u*)(p + 16);
  return f.v;
}
// B fragment from LDS tile (row-major [64][BROW], rows are N, 32 K-elems per row).
// lane<16: N=lane, K = {0..7, 16..23}; lane>=16: +8 on K base.
__device__ __forceinline__ v16bf load_frag_B_lds(const bf16_t* base) {
  ABFrag f;
  f.q[0] = *(const v4u*)(base);
  f.q[1] = *(const v4u*)(base + 16);
  return f.v;
}

// Async DMA of one 16B segment of the 64x32 B tile into LDS (gfx1250 async path,
// tracked by ASYNCcnt).  256 threads x 16B = full 4KB tile.
__device__ __forceinline__ void stage_async(const bf16_t* W, int ldb, int n0, int k0,
                                            bf16_t* dstSm, int srow, int sseg) {
  const bf16_t* g = W + (size_t)(n0 + srow) * ldb + k0 + sseg;
  unsigned lds = (unsigned)(unsigned long long)(const void*)(dstSm + srow * BROW + sseg);
  unsigned long long ga = (unsigned long long)g;
  asm volatile("global_load_async_to_lds_b128 %0, %1, off" :: "v"(lds), "v"(ga) : "memory");
}
__device__ __forceinline__ void wait_async0() {
  asm volatile("s_wait_asynccnt 0" ::: "memory");
}

// ------------------------------------------------------------------
// Fused GEMM: C[m][n] = sum_k A1[m][k] W1[n][k]  (+ sum_k A2[m][k] W2[n][k]) + bias[n]
// Block = 256 thr = 8 waves; block tile 128(M) x 64(N); wave tile 16x64.
// grid.x = N/64 (N padded weight rows to mult of 64), grid.y = M/128.
// K1,K2 multiples of 32.  B tile double-buffered in LDS via async DMA.
// ------------------------------------------------------------------
__global__ __launch_bounds__(256) void k_gemm(
    const bf16_t* __restrict__ A1, int lda1, const bf16_t* __restrict__ W1, int ldb1, int K1,
    const bf16_t* __restrict__ A2, int lda2, const bf16_t* __restrict__ W2, int ldb2, int K2,
    const float* __restrict__ bias,
    float* __restrict__ C, int ldc,
    bf16_t* __restrict__ Cbf, int ldcbf,
    int N, int relu)
{
  __shared__ bf16_t Bsm[2][64 * BROW];

  const int tid  = threadIdx.x;
  const int wave = tid >> 5;
  const int lane = tid & 31;
  const int hf   = lane >> 4;
  const int lo   = lane & 15;
  const int m0   = blockIdx.y * 128 + wave * 16;
  const int n0   = blockIdx.x * 64;
  const int mrow = m0 + lo;
  const int srow = tid >> 2;          // 0..63 : B-tile row this thread stages
  const int sseg = (tid & 3) * 8;     // 0,8,16,24 : element offset within row

  v8f acc[4] = {};

  const bf16_t* As[2]   = {A1, A2};
  const bf16_t* Ws[2]   = {W1, W2};
  const int     ldas[2] = {lda1, lda2};
  const int     ldbs[2] = {ldb1, ldb2};
  const int     Ks[2]   = {K1, K2};

  for (int p = 0; p < 2; ++p) {
    const bf16_t* A = As[p];
    if (A == nullptr) continue;
    const bf16_t* W = Ws[p];
    const int lda = ldas[p], ldb = ldbs[p], K = Ks[p];

    __syncthreads();                               // protect Bsm reuse across pairs
    stage_async(W, ldb, n0, 0, Bsm[0], srow, sseg);
    int cur = 0;

    for (int k0 = 0; k0 < K; k0 += 32) {
      wait_async0();                               // my tile segment landed
      __syncthreads();                             // everyone's segment landed
      if (k0 + 32 < K)                             // overlap next DMA with math
        stage_async(W, ldb, n0, k0 + 32, Bsm[cur ^ 1], srow, sseg);

      if (k0 + 32 < K)                             // prefetch next A tile
        __builtin_prefetch(A + (size_t)mrow * lda + k0 + 32, 0, 1);

      v16bf a = load_frag_A(A, lda, mrow, k0, hf);
      const bf16_t* bs = &Bsm[cur][(size_t)lo * BROW + hf * 8];
#pragma unroll
      for (int f = 0; f < 4; ++f) {
        v16bf b = load_frag_B_lds(bs + f * 16 * BROW);
        acc[f] = __builtin_amdgcn_wmma_f32_16x16x32_bf16(
            false, a, false, b, (short)0, acc[f], false, false);
      }
      cur ^= 1;
    }
  }
  __syncthreads();

#pragma unroll
  for (int f = 0; f < 4; ++f) {
    int ncol = n0 + f * 16 + lo;
    bool valid = ncol < N;
    float bv = (bias != nullptr && valid) ? bias[ncol] : 0.0f;
#pragma unroll
    for (int j = 0; j < 8; ++j) {
      int row = m0 + hf * 8 + j;                // C/D layout: VGPR j -> M=j (+8 for hi half)
      float v = acc[f][j] + bv;
      if (relu) v = v > 0.0f ? v : 0.0f;
      if (valid) {
        C[(size_t)row * ldc + ncol] = v;
        if (Cbf) Cbf[(size_t)row * ldcbf + ncol] = f2bf(v);
      }
    }
  }
}

// ---------------- conversion / init ----------------
__global__ void k_cvt_pad(const float* __restrict__ in, int rows, int cols,
                          bf16_t* __restrict__ out, int colsPad, long total)
{
  long i = (long)blockIdx.x * blockDim.x + threadIdx.x;
  if (i >= total) return;
  int  c = (int)(i % colsPad);
  long r = i / colsPad;
  float v = (r < rows && c < cols) ? in[r * (long)cols + c] : 0.0f;
  out[i] = f2bf(v);
}

__global__ void k_zero_f32(float* p, long n) {
  long i = (long)blockIdx.x * blockDim.x + threadIdx.x;
  if (i < n) p[i] = 0.0f;
}
__global__ void k_zero_bf(bf16_t* p, long n) {
  long i = (long)blockIdx.x * blockDim.x + threadIdx.x;
  if (i < n) p[i] = __builtin_bit_cast(bf16_t, (unsigned short)0);
}

// ---------------- LSTM cells ----------------
__global__ __launch_bounds__(256) void k_cell_enc(
    const float* __restrict__ z, float* __restrict__ c, bf16_t* __restrict__ hbf,
    float* __restrict__ raw, const int* __restrict__ lengths, int t)
{
  int i = blockIdx.x * 256 + threadIdx.x;
  if (i >= BB * HH) return;
  int b = i >> 9, h = i & (HH - 1);
  const float* zr = z + (size_t)b * G4H;
  float zi = zr[h], zf = zr[HH + h], zg = zr[2 * HH + h], zo = zr[3 * HH + h];
  float cn = sigm(zf) * c[i] + sigm(zi) * tanhf(zg);
  float hn = sigm(zo) * tanhf(cn);
  c[i]   = cn;
  hbf[i] = f2bf(hn);
  raw[((size_t)t * BB + b) * HH + h] = (t < lengths[b]) ? hn : 0.0f;
}

__global__ __launch_bounds__(256) void k_cell_dec(
    const float* __restrict__ z, float* __restrict__ c, bf16_t* __restrict__ dhbf,
    float* __restrict__ din_f, bf16_t* __restrict__ din_b)
{
  int i = blockIdx.x * 256 + threadIdx.x;
  if (i >= BB * HH) return;
  int b = i >> 9, h = i & (HH - 1);
  const float* zr = z + (size_t)b * G4H;
  float zi = zr[h], zf = zr[HH + h], zg = zr[2 * HH + h], zo = zr[3 * HH + h];
  float cn = sigm(zf) * c[i] + sigm(zi) * tanhf(zg);
  float hn = sigm(zo) * tanhf(cn);
  c[i]    = cn;
  dhbf[i] = f2bf(hn);
  din_f[(size_t)b * H2 + h] = hn;
  din_b[(size_t)b * H2 + h] = f2bf(hn);
}

// ---------------- BatchNorm (training-mode batch stats) ----------------
__global__ __launch_bounds__(256) void k_bn_stats(
    const float* __restrict__ raw, float* __restrict__ mu, float* __restrict__ rs)
{
  __shared__ float s1[256], s2[256];
  int ch = blockIdx.x, tid = threadIdx.x;
  float a = 0.0f, b = 0.0f;
  for (int i = tid; i < SS * BB; i += 256) {
    float x = raw[(size_t)i * HH + ch];
    a += x; b += x * x;
  }
  s1[tid] = a; s2[tid] = b; __syncthreads();
  for (int o = 128; o > 0; o >>= 1) {
    if (tid < o) { s1[tid] += s1[tid + o]; s2[tid] += s2[tid + o]; }
    __syncthreads();
  }
  if (tid == 0) {
    float m = s1[0] / (float)(SS * BB);
    float v = s2[0] / (float)(SS * BB) - m * m;
    mu[ch] = m;
    rs[ch] = rsqrtf(v + 1e-5f);
  }
}

__global__ __launch_bounds__(256) void k_bn_apply(
    const float* __restrict__ raw, const float* __restrict__ gamma,
    const float* __restrict__ beta, const float* __restrict__ mu,
    const float* __restrict__ rs, bf16_t* __restrict__ out, long n)
{
  long i = (long)blockIdx.x * blockDim.x + threadIdx.x;
  if (i >= n) return;
  int ch = (int)(i & (HH - 1));
  out[i] = f2bf(gamma[ch] * (raw[i] - mu[ch]) * rs[ch] + beta[ch]);
}

// ---------------- attention scores: one wave per (s,b) row ----------------
__global__ __launch_bounds__(256) void k_scores(
    const float* __restrict__ encProj, const float* __restrict__ dpb,
    const float* __restrict__ w2, float* __restrict__ scores)
{
  int gw = blockIdx.x * 8 + (threadIdx.x >> 5);   // gw = s*128 + b
  int lane = threadIdx.x & 31;
  if (gw >= SS * BB) return;
  int b = gw & (BB - 1);
  const float* ep = encProj + (size_t)gw * HH;
  const float* dp = dpb + (size_t)b * HH;
  float acc = 0.0f;
  for (int d = lane; d < HH; d += 32) acc += tanhf(ep[d] + dp[d]) * w2[d];
  for (int o = 16; o > 0; o >>= 1) acc += __shfl_xor(acc, o, 32);
  if (lane == 0) scores[gw] = acc;
}

// ---------------- softmax (multiplicative mask) + context + attn output ----------------
__global__ __launch_bounds__(256) void k_softmax_ctx(
    const float* __restrict__ scores, const int* __restrict__ lengths,
    const bf16_t* __restrict__ enc_hs, float* __restrict__ attn_out, int tstep,
    float* __restrict__ din_f, bf16_t* __restrict__ din_b)
{
  __shared__ float red[256];
  __shared__ float w[256];
  int b = blockIdx.x, s = threadIdx.x;
  int len = lengths[b];
  float sc = scores[(size_t)s * BB + b];
  sc = (s < len) ? sc : 0.0f;                    // mask * scores (multiplicative)
  red[s] = sc; __syncthreads();
  for (int o = 128; o > 0; o >>= 1) {
    if (s < o) red[s] = fmaxf(red[s], red[s + o]);
    __syncthreads();
  }
  float mx = red[0]; __syncthreads();
  float e = expf(sc - mx);
  red[s] = e; __syncthreads();
  for (int o = 128; o > 0; o >>= 1) {
    if (s < o) red[s] += red[s + o];
    __syncthreads();
  }
  float wv = e / red[0];
  w[s] = wv; __syncthreads();

  if (attn_out != nullptr) {
    float o = ((tstep < len) && (s < len)) ? wv : 1e-10f;
    attn_out[((size_t)b * SS + tstep) * SS + s] = o;
  }
  // ctx[b][h] = sum_s w[s] * enc_hs[s][b][h]  -> din[b][H + h]
  for (int h = threadIdx.x; h < HH; h += 256) {
    float acc = 0.0f;
    for (int s2 = 0; s2 < SS; ++s2)
      acc += w[s2] * bf2f(enc_hs[((size_t)s2 * BB + b) * HH + h]);
    din_f[(size_t)b * H2 + HH + h] = acc;
    din_b[(size_t)b * H2 + HH + h] = f2bf(acc);
  }
}

// ---------------- final masked store of decoder output ----------------
__global__ __launch_bounds__(256) void k_store_out(
    const float* __restrict__ fc2, const int* __restrict__ lengths,
    float* __restrict__ out_dec, int t)
{
  int i = blockIdx.x * 256 + threadIdx.x;
  if (i >= BB * TGTN) return;
  int b = i / TGTN, j = i - b * TGTN;
  out_dec[((size_t)b * SS + t) * TGTN + j] =
      (t < lengths[b]) ? fc2[(size_t)b * TGTP + j] : 0.0f;
}

// ==================================================================
extern "C" void kernel_launch(void* const* d_in, const int* in_sizes, int n_in,
                              void* d_out, int out_size, void* d_ws, size_t ws_size,
                              hipStream_t stream) {
  (void)in_sizes; (void)n_in; (void)out_size; (void)ws_size;

  const float* source   = (const float*)d_in[0];
  const int*   lengths  = (const int*)d_in[2];
  const float* W_ih_enc = (const float*)d_in[3];
  const float* W_hh_enc = (const float*)d_in[4];
  const float* b_enc    = (const float*)d_in[5];
  const float* gamma    = (const float*)d_in[6];
  const float* beta     = (const float*)d_in[7];
  const float* W_ih_dec = (const float*)d_in[8];
  const float* W_hh_dec = (const float*)d_in[9];
  const float* b_dec    = (const float*)d_in[10];
  const float* W_attn1  = (const float*)d_in[11];
  const float* b_attn1  = (const float*)d_in[12];
  const float* w_attn2  = (const float*)d_in[13];
  const float* W_fc1    = (const float*)d_in[14];
  const float* b_fc1    = (const float*)d_in[15];
  const float* W_fc2    = (const float*)d_in[16];
  const float* b_fc2    = (const float*)d_in[17];

  float* out_dec  = (float*)d_out;                               // [B, S, TGT]
  float* out_attn = (float*)d_out + (size_t)BB * SS * TGTN;      // [B, S, S]

  // ---- workspace carve-out ----
  char* ws = (char*)d_ws;
  size_t off = 0;
  auto alloc = [&](size_t bytes) -> void* {
    void* p = ws + off;
    off += (bytes + 255) & ~(size_t)255;
    return p;
  };
  bf16_t* src_bf     = (bf16_t*)alloc((size_t)BB * SS * SRCP * 2);
  bf16_t* Wih_enc_bf = (bf16_t*)alloc((size_t)G4H * SRCP * 2);
  bf16_t* Whh_enc_bf = (bf16_t*)alloc((size_t)G4H * HH * 2);
  bf16_t* Wih_dec_bf = (bf16_t*)alloc((size_t)G4H * H2 * 2);
  bf16_t* Whh_dec_bf = (bf16_t*)alloc((size_t)G4H * HH * 2);
  bf16_t* Wattn1_bf  = (bf16_t*)alloc((size_t)HH * H2 * 2);
  bf16_t* Wfc1_bf    = (bf16_t*)alloc((size_t)H2 * H2 * 2);
  bf16_t* Wfc2_bf    = (bf16_t*)alloc((size_t)TGTP * H2 * 2);
  float*  enc_raw    = (float*)alloc((size_t)SS * BB * HH * 4);  // later reused as encProj
  bf16_t* enc_hs_bf  = (bf16_t*)alloc((size_t)SS * BB * HH * 2);
  float*  mu         = (float*)alloc(HH * 4);
  float*  rs         = (float*)alloc(HH * 4);
  bf16_t* h_enc_bf   = (bf16_t*)alloc((size_t)BB * HH * 2);
  float*  c_enc      = (float*)alloc((size_t)BB * HH * 4);
  float*  c_dec      = (float*)alloc((size_t)BB * HH * 4);
  bf16_t* dh_bf      = (bf16_t*)alloc((size_t)BB * HH * 2);
  float*  zbuf       = (float*)alloc((size_t)BB * G4H * 4);
  float*  din_f      = (float*)alloc((size_t)BB * H2 * 4);
  bf16_t* din_b      = (bf16_t*)alloc((size_t)BB * H2 * 2);
  float*  dec_projb  = (float*)alloc((size_t)BB * HH * 4);
  float*  scores     = (float*)alloc((size_t)SS * BB * 4);
  float*  fc1_f      = (float*)alloc((size_t)BB * H2 * 4);
  bf16_t* fc1_bf     = (bf16_t*)alloc((size_t)BB * H2 * 2);
  float*  fc2_f      = (float*)alloc((size_t)BB * TGTP * 4);
  float*  encProj    = enc_raw;   // reuse after BN apply

  auto cvt = [&](const float* in, int rows, int cols, bf16_t* out, int colsPad, int rowsPad) {
    long total = (long)rowsPad * colsPad;
    k_cvt_pad<<<dim3((unsigned)((total + 255) / 256)), dim3(256), 0, stream>>>(
        in, rows, cols, out, colsPad, total);
  };

  // ---- conversions (bf16, padded) ----
  cvt(source,   BB * SS, SRCN, src_bf,     SRCP, BB * SS);
  cvt(W_ih_enc, G4H,     SRCN, Wih_enc_bf, SRCP, G4H);
  cvt(W_hh_enc, G4H,     HH,   Whh_enc_bf, HH,   G4H);
  cvt(W_ih_dec, G4H,     H2,   Wih_dec_bf, H2,   G4H);
  cvt(W_hh_dec, G4H,     HH,   Whh_dec_bf, HH,   G4H);
  cvt(W_attn1,  HH,      H2,   Wattn1_bf,  H2,   HH);
  cvt(W_fc1,    H2,      H2,   Wfc1_bf,    H2,   H2);
  cvt(W_fc2,    TGTN,    H2,   Wfc2_bf,    H2,   TGTP);

  // ---- zero-init state ----
  k_zero_bf <<<dim3(256), dim3(256), 0, stream>>>(h_enc_bf, (long)BB * HH);
  k_zero_f32<<<dim3(256), dim3(256), 0, stream>>>(c_enc,    (long)BB * HH);
  k_zero_f32<<<dim3(256), dim3(256), 0, stream>>>(c_dec,    (long)BB * HH);
  k_zero_bf <<<dim3(256), dim3(256), 0, stream>>>(dh_bf,    (long)BB * HH);
  k_zero_f32<<<dim3(512), dim3(256), 0, stream>>>(din_f,    (long)BB * H2);
  k_zero_bf <<<dim3(512), dim3(256), 0, stream>>>(din_b,    (long)BB * H2);

  // ---- encoder recurrence: z = x_t@Wih^T + h@Whh^T + b ----
  for (int t = 0; t < SS; ++t) {
    k_gemm<<<dim3(G4H / 64, 1), dim3(256), 0, stream>>>(
        src_bf + (size_t)t * SRCP, SS * SRCP, Wih_enc_bf, SRCP, SRCP,
        h_enc_bf, HH, Whh_enc_bf, HH, HH,
        b_enc, zbuf, G4H, (bf16_t*)nullptr, 0, G4H, 0);
    k_cell_enc<<<dim3(BB * HH / 256), dim3(256), 0, stream>>>(
        zbuf, c_enc, h_enc_bf, enc_raw, lengths, t);
  }

  // ---- BatchNorm over (S,B) per channel ----
  k_bn_stats<<<dim3(HH), dim3(256), 0, stream>>>(enc_raw, mu, rs);
  k_bn_apply<<<dim3((unsigned)(((long)SS * BB * HH + 255) / 256)), dim3(256), 0, stream>>>(
      enc_raw, gamma, beta, mu, rs, enc_hs_bf, (long)SS * BB * HH);

  // ---- enc_proj = enc_hs @ W_attn1[:, :H]^T   (overwrites enc_raw) ----
  k_gemm<<<dim3(HH / 64, (SS * BB) / 128), dim3(256), 0, stream>>>(
      enc_hs_bf, HH, Wattn1_bf, H2, HH,
      (bf16_t*)nullptr, 0, (bf16_t*)nullptr, 0, 0,
      (const float*)nullptr, encProj, HH, (bf16_t*)nullptr, 0, HH, 0);

  // ---- initial attend with h0 = 0 (dec_proj = bias only) ----
  k_gemm<<<dim3(HH / 64, 1), dim3(256), 0, stream>>>(
      dh_bf, HH, Wattn1_bf + HH, H2, HH,
      (bf16_t*)nullptr, 0, (bf16_t*)nullptr, 0, 0,
      b_attn1, dec_projb, HH, (bf16_t*)nullptr, 0, HH, 0);
  k_scores<<<dim3(SS * BB / 8), dim3(256), 0, stream>>>(encProj, dec_projb, w_attn2, scores);
  k_softmax_ctx<<<dim3(BB), dim3(256), 0, stream>>>(
      scores, lengths, enc_hs_bf, (float*)nullptr, -1, din_f, din_b);

  // ---- decoder steps ----
  for (int t = 0; t < SS; ++t) {
    // LSTM: z = din@Wih_dec^T + dh@Whh_dec^T + b
    k_gemm<<<dim3(G4H / 64, 1), dim3(256), 0, stream>>>(
        din_b, H2, Wih_dec_bf, H2, H2,
        dh_bf, HH, Whh_dec_bf, HH, HH,
        b_dec, zbuf, G4H, (bf16_t*)nullptr, 0, G4H, 0);
    k_cell_dec<<<dim3(BB * HH / 256), dim3(256), 0, stream>>>(
        zbuf, c_dec, dh_bf, din_f, din_b);
    // attention
    k_gemm<<<dim3(HH / 64, 1), dim3(256), 0, stream>>>(
        dh_bf, HH, Wattn1_bf + HH, H2, HH,
        (bf16_t*)nullptr, 0, (bf16_t*)nullptr, 0, 0,
        b_attn1, dec_projb, HH, (bf16_t*)nullptr, 0, HH, 0);
    k_scores<<<dim3(SS * BB / 8), dim3(256), 0, stream>>>(encProj, dec_projb, w_attn2, scores);
    k_softmax_ctx<<<dim3(BB), dim3(256), 0, stream>>>(
        scores, lengths, enc_hs_bf, out_attn, t, din_f, din_b);
    // fc1 (relu, bf16 copy for fc2 input)
    k_gemm<<<dim3(H2 / 64, 1), dim3(256), 0, stream>>>(
        din_b, H2, Wfc1_bf, H2, H2,
        (bf16_t*)nullptr, 0, (bf16_t*)nullptr, 0, 0,
        b_fc1, fc1_f, H2, fc1_bf, H2, H2, 1);
    // fc2 (N=300, weights padded to 320 rows)
    k_gemm<<<dim3(TGTP / 64, 1), dim3(256), 0, stream>>>(
        fc1_bf, H2, Wfc2_bf, H2, H2,
        (bf16_t*)nullptr, 0, (bf16_t*)nullptr, 0, 0,
        b_fc2, fc2_f, TGTP, (bf16_t*)nullptr, 0, TGTN, 0);
    k_store_out<<<dim3((BB * TGTN + 255) / 256), dim3(256), 0, stream>>>(
        fc2_f, lengths, out_dec, t);
  }
}
